// GAT_11046655886004
// MI455X (gfx1250) — matrix operand order repaired and checked by reading, not compiled, not run on hardware
//
#include <hip/hip_runtime.h>
#include <hip/hip_bf16.h>

// ---------------------------------------------------------------------------
// 2-layer GAT for MI455X (gfx1250, wave32).
//   GEMMs: v_wmma_f32_16x16x32_bf16, weights staged in LDS as bf16,
//          32 output rows per wave (2 A-frags x 4 N-tiles = 8 WMMAs / K-step).
//   Edge softmax/aggregation: float4 L2-resident gathers + f32/u32 atomics.
// ---------------------------------------------------------------------------

#define NNODES 50000
#define NEDGES 1600000
#define FIN    512
#define HEADS  8
#define DHEAD  8
#define C1     64            // HEADS*DHEAD
#define NCLS   16
#define NEG_SLOPE 0.2f
#define GAT_EPS 1e-16f

typedef __attribute__((ext_vector_type(16))) __bf16 v16bf;
typedef __attribute__((ext_vector_type(8)))  float  v8f;

union F8  { float4 v[2]; float f[8];  };
union F16 { float4 v[4]; float f[16]; };

__device__ __forceinline__ unsigned short f2bf_u(float f) {
  unsigned u = __float_as_uint(f);
  u += 0x7FFFu + ((u >> 16) & 1u);           // round-to-nearest-even
  return (unsigned short)(u >> 16);
}
__device__ __forceinline__ unsigned ordf(float f) {
  unsigned u = __float_as_uint(f);
  return (u & 0x80000000u) ? ~u : (u | 0x80000000u);   // order-preserving map
}
__device__ __forceinline__ float dec_ord(unsigned e) {
  return (e & 0x80000000u) ? __uint_as_float(e & 0x7FFFFFFFu)
                           : __uint_as_float(~e);       // e==0 -> NaN (no edges)
}
__device__ __forceinline__ float lrelu(float x) { return x > 0.f ? x : NEG_SLOPE * x; }

__device__ __forceinline__ F8 load8(const float* __restrict__ p) {
  F8 r;
  r.v[0] = *(const float4*)p;
  r.v[1] = *(const float4*)(p + 4);
  return r;
}

// A fragment 16x32 bf16 (ISA layout): lane half 0 holds K {b..b+7, b+16..b+23},
// half 1 holds K {b+8..b+15, b+24..b+31}; two contiguous 32B f32 runs per lane.
__device__ __forceinline__ v16bf load_a_frag(const float* __restrict__ xrow,
                                             int ks, int half) {
  const float* p = xrow + ks * 32 + half * 8;
  float4 q0 = *(const float4*)(p);
  float4 q1 = *(const float4*)(p + 4);
  float4 q2 = *(const float4*)(p + 16);
  float4 q3 = *(const float4*)(p + 20);
  union { unsigned short s[16]; v16bf v; } au;
  au.s[0]=f2bf_u(q0.x);  au.s[1]=f2bf_u(q0.y);  au.s[2]=f2bf_u(q0.z);  au.s[3]=f2bf_u(q0.w);
  au.s[4]=f2bf_u(q1.x);  au.s[5]=f2bf_u(q1.y);  au.s[6]=f2bf_u(q1.z);  au.s[7]=f2bf_u(q1.w);
  au.s[8]=f2bf_u(q2.x);  au.s[9]=f2bf_u(q2.y);  au.s[10]=f2bf_u(q2.z); au.s[11]=f2bf_u(q2.w);
  au.s[12]=f2bf_u(q3.x); au.s[13]=f2bf_u(q3.y); au.s[14]=f2bf_u(q3.z); au.s[15]=f2bf_u(q3.w);
  return au.v;
}

// ---------------------------------------------------------------------------
// GEMM1: H1[N,64] = X[N,512] @ W1[512,64].
// Block = 256 threads = 8 waves; wave computes a 32x64 strip.
// ---------------------------------------------------------------------------
__global__ __launch_bounds__(256) void k_gemm1(const float* __restrict__ X,
                                               const float* __restrict__ W1,
                                               float* __restrict__ H1) {
  __shared__ unsigned short w1s[FIN * C1];   // 64 KB bf16 copy of W1
  int tid = threadIdx.x;
  for (int i = tid; i < FIN * C1; i += 256) w1s[(i % C1) * FIN + i / C1] = f2bf_u(W1[i]);
  __syncthreads();

  int wave = tid >> 5, lane = tid & 31;
  int half = lane >> 4, l15 = lane & 15;
  int mbase = blockIdx.x * 256 + wave * 32;
  int m0 = mbase + l15;
  int m1 = m0 + 16;
  int m0c = m0 < NNODES ? m0 : NNODES - 1;
  int m1c = m1 < NNODES ? m1 : NNODES - 1;
  const float* xr0 = X + (size_t)m0c * FIN;
  const float* xr1 = X + (size_t)m1c * FIN;

  v8f acc[2][4] = {};
  for (int ks = 0; ks < FIN / 32; ++ks) {
    v16bf a0 = load_a_frag(xr0, ks, half);
    v16bf a1 = load_a_frag(xr1, ks, half);
    // B fragment 32x16 bf16: lane = N col, K in A's table order (2x ds_load_b128)
#pragma unroll
    for (int t = 0; t < 4; ++t) {
      union { uint4 u[2]; v16bf v; } bu;
      const uint4* bp = (const uint4*)&w1s[(t * 16 + l15) * FIN + ks * 32 + half * 8];
      bu.u[0] = bp[0];
      bu.u[1] = bp[2];
      acc[0][t] = __builtin_amdgcn_wmma_f32_16x16x32_bf16(
          false, a0, false, bu.v, (short)0, acc[0][t], false, false);
      acc[1][t] = __builtin_amdgcn_wmma_f32_16x16x32_bf16(
          false, a1, false, bu.v, (short)0, acc[1][t], false, false);
    }
  }

  // C/D layout: VGPR r holds row M = g*16 + half*8 + r, col N = t*16 + (lane&15)
  if (mbase + 32 <= NNODES) {            // wave-uniform fast path: no guards
    float* op = H1 + (size_t)(mbase + half * 8) * C1 + l15;
#pragma unroll
    for (int g = 0; g < 2; ++g)
#pragma unroll
      for (int t = 0; t < 4; ++t)
#pragma unroll
        for (int r = 0; r < 8; ++r)
          op[(size_t)(g * 16 + r) * C1 + t * 16] = acc[g][t][r];
  } else {
#pragma unroll
    for (int g = 0; g < 2; ++g)
#pragma unroll
      for (int t = 0; t < 4; ++t)
#pragma unroll
        for (int r = 0; r < 8; ++r) {
          int row = mbase + g * 16 + half * 8 + r;
          if (row < NNODES) H1[(size_t)row * C1 + t * 16 + l15] = acc[g][t][r];
        }
  }
}

// ---------------------------------------------------------------------------
// GEMM2: H2[N,16] = Hact[N,64] @ W2[64,16]  (2 K-steps, 32 rows per wave).
// ---------------------------------------------------------------------------
__global__ __launch_bounds__(256) void k_gemm2(const float* __restrict__ Hact,
                                               const float* __restrict__ W2,
                                               float* __restrict__ H2) {
  __shared__ unsigned short w2s[C1 * NCLS];  // 2 KB
  int tid = threadIdx.x;
  for (int i = tid; i < C1 * NCLS; i += 256) w2s[(i % NCLS) * C1 + i / NCLS] = f2bf_u(W2[i]);
  __syncthreads();

  int wave = tid >> 5, lane = tid & 31;
  int half = lane >> 4, l15 = lane & 15;
  int mbase = blockIdx.x * 256 + wave * 32;
  int m0 = mbase + l15;
  int m1 = m0 + 16;
  int m0c = m0 < NNODES ? m0 : NNODES - 1;
  int m1c = m1 < NNODES ? m1 : NNODES - 1;
  const float* hr0 = Hact + (size_t)m0c * C1;
  const float* hr1 = Hact + (size_t)m1c * C1;

  v8f acc[2] = {};
#pragma unroll
  for (int ks = 0; ks < 2; ++ks) {
    v16bf a0 = load_a_frag(hr0, ks, half);
    v16bf a1 = load_a_frag(hr1, ks, half);
    union { uint4 u[2]; v16bf v; } bu;
    const uint4* bp = (const uint4*)&w2s[l15 * C1 + ks * 32 + half * 8];
    bu.u[0] = bp[0];
    bu.u[1] = bp[2];
    acc[0] = __builtin_amdgcn_wmma_f32_16x16x32_bf16(
        false, a0, false, bu.v, (short)0, acc[0], false, false);
    acc[1] = __builtin_amdgcn_wmma_f32_16x16x32_bf16(
        false, a1, false, bu.v, (short)0, acc[1], false, false);
  }

  if (mbase + 32 <= NNODES) {
    float* op = H2 + (size_t)(mbase + half * 8) * NCLS + l15;
#pragma unroll
    for (int g = 0; g < 2; ++g)
#pragma unroll
      for (int r = 0; r < 8; ++r)
        op[(size_t)(g * 16 + r) * NCLS] = acc[g][r];
  } else {
#pragma unroll
    for (int g = 0; g < 2; ++g)
#pragma unroll
      for (int r = 0; r < 8; ++r) {
        int row = mbase + g * 16 + half * 8 + r;
        if (row < NNODES) H2[(size_t)row * NCLS + l15] = acc[g][r];
      }
  }
}

// ---------------------------------------------------------------------------
// Elementwise / edge kernels
// ---------------------------------------------------------------------------
__global__ void k_zero4(float4* __restrict__ p, int n4) {
  int i = blockIdx.x * blockDim.x + threadIdx.x;
  if (i < n4) p[i] = make_float4(0.f, 0.f, 0.f, 0.f);
}

__global__ void k_scores1(const float* __restrict__ H1, const float* __restrict__ ats,
                          const float* __restrict__ atd, float* __restrict__ as,
                          float* __restrict__ ad) {
  int i = blockIdx.x * blockDim.x + threadIdx.x;
  if (i >= NNODES * HEADS) return;
  int n = i >> 3, h = i & 7;
  F8 hp = load8(H1 + (size_t)n * C1 + h * DHEAD);
  F8 sp = load8(ats + h * DHEAD);
  F8 dp = load8(atd + h * DHEAD);
  float vs = 0.f, vd = 0.f;
#pragma unroll
  for (int c = 0; c < DHEAD; ++c) { vs += hp.f[c] * sp.f[c]; vd += hp.f[c] * dp.f[c]; }
  as[i] = vs; ad[i] = vd;
}

__global__ void k_edge_max1(const int* __restrict__ src, const int* __restrict__ dst,
                            const float* __restrict__ as, const float* __restrict__ ad,
                            unsigned* __restrict__ mraw) {
  int e = blockIdx.x * blockDim.x + threadIdx.x;
  if (e >= NEDGES) return;
  int s = src[e], d = dst[e];
  F8 es = load8(as + (size_t)s * HEADS);
  F8 ed = load8(ad + (size_t)d * HEADS);
#pragma unroll
  for (int h = 0; h < HEADS; ++h)
    atomicMax(&mraw[(size_t)d * HEADS + h], ordf(lrelu(es.f[h] + ed.f[h])));
}

__global__ void k_node1(const float* __restrict__ as, const float* __restrict__ ad,
                        unsigned* __restrict__ mraw, float* __restrict__ s1) {
  int i = blockIdx.x * blockDim.x + threadIdx.x;
  if (i >= NNODES * HEADS) return;
  float self = lrelu(as[i] + ad[i]);
  float m = fmaxf(dec_ord(mraw[i]), self);       // fmaxf drops the no-edge NaN
  ((float*)mraw)[i] = m;                         // buffer becomes float m from now on
  s1[i] = __expf(self - m);                      // self-loop contribution to the sum
}

__global__ void k_edge_sum1(const int* __restrict__ src, const int* __restrict__ dst,
                            const float* __restrict__ as, const float* __restrict__ ad,
                            const float* __restrict__ mf, float* __restrict__ s1) {
  int e = blockIdx.x * blockDim.x + threadIdx.x;
  if (e >= NEDGES) return;
  int s = src[e], d = dst[e];
  F8 es = load8(as + (size_t)s * HEADS);
  F8 ed = load8(ad + (size_t)d * HEADS);
  F8 md = load8(mf + (size_t)d * HEADS);
#pragma unroll
  for (int h = 0; h < HEADS; ++h) {
    float al = lrelu(es.f[h] + ed.f[h]);
    atomicAdd(&s1[(size_t)d * HEADS + h], __expf(al - md.f[h]));
  }
}

__global__ void k_edge_agg1(const int* __restrict__ src, const int* __restrict__ dst,
                            const float* __restrict__ as, const float* __restrict__ ad,
                            const float* __restrict__ mf, const float* __restrict__ s1,
                            const float* __restrict__ H1, float* __restrict__ OUT1) {
  int i = blockIdx.x * blockDim.x + threadIdx.x;   // over E*HEADS
  if (i >= NEDGES * HEADS) return;
  int e = i >> 3, h = i & 7;
  int s = src[e], d = dst[e];
  float al = lrelu(as[(size_t)s * HEADS + h] + ad[(size_t)d * HEADS + h]);
  float coef = __expf(al - mf[(size_t)d * HEADS + h]) /
               (s1[(size_t)d * HEADS + h] + GAT_EPS);
  F8 hp = load8(H1 + (size_t)s * C1 + h * DHEAD);
  float* op = OUT1 + (size_t)d * C1 + h * DHEAD;
#pragma unroll
  for (int c = 0; c < DHEAD; ++c) atomicAdd(op + c, hp.f[c] * coef);
}

__global__ void k_out1(const float* __restrict__ as, const float* __restrict__ ad,
                       const float* __restrict__ mf, const float* __restrict__ s1,
                       const float* __restrict__ H1, const float* __restrict__ bias1,
                       float* __restrict__ OUT1) {
  int i = blockIdx.x * blockDim.x + threadIdx.x;   // over N*C1
  if (i >= NNODES * C1) return;
  int n = i >> 6, c = i & 63, h = c >> 3;
  float self = lrelu(as[n * HEADS + h] + ad[n * HEADS + h]);
  float coef = __expf(self - mf[n * HEADS + h]) / (s1[n * HEADS + h] + GAT_EPS);
  float v = OUT1[i] + H1[i] * coef + bias1[c];
  OUT1[i] = v > 0.f ? v : __expf(v) - 1.f;         // ELU, in place
}

__global__ void k_scores2(const float* __restrict__ H2, const float* __restrict__ ats,
                          const float* __restrict__ atd, float* __restrict__ as,
                          float* __restrict__ ad) {
  int n = blockIdx.x * blockDim.x + threadIdx.x;
  if (n >= NNODES) return;
  F16 hp; F16 sp; F16 dp;
#pragma unroll
  for (int q = 0; q < 4; ++q) {
    hp.v[q] = *(const float4*)(H2 + (size_t)n * NCLS + q * 4);
    sp.v[q] = *(const float4*)(ats + q * 4);
    dp.v[q] = *(const float4*)(atd + q * 4);
  }
  float vs = 0.f, vd = 0.f;
#pragma unroll
  for (int c = 0; c < NCLS; ++c) { vs += hp.f[c] * sp.f[c]; vd += hp.f[c] * dp.f[c]; }
  as[n] = vs; ad[n] = vd;
}

__global__ void k_edge_max2(const int* __restrict__ src, const int* __restrict__ dst,
                            const float* __restrict__ as, const float* __restrict__ ad,
                            unsigned* __restrict__ mraw) {
  int e = blockIdx.x * blockDim.x + threadIdx.x;
  if (e >= NEDGES) return;
  float al = lrelu(as[src[e]] + ad[dst[e]]);
  atomicMax(&mraw[dst[e]], ordf(al));
}

__global__ void k_node2(const float* __restrict__ as, const float* __restrict__ ad,
                        unsigned* __restrict__ mraw, float* __restrict__ s2) {
  int n = blockIdx.x * blockDim.x + threadIdx.x;
  if (n >= NNODES) return;
  float self = lrelu(as[n] + ad[n]);
  float m = fmaxf(dec_ord(mraw[n]), self);
  ((float*)mraw)[n] = m;
  s2[n] = __expf(self - m);
}

__global__ void k_edge_sum2(const int* __restrict__ src, const int* __restrict__ dst,
                            const float* __restrict__ as, const float* __restrict__ ad,
                            const float* __restrict__ mf, float* __restrict__ s2) {
  int e = blockIdx.x * blockDim.x + threadIdx.x;
  if (e >= NEDGES) return;
  int d = dst[e];
  float al = lrelu(as[src[e]] + ad[d]);
  atomicAdd(&s2[d], __expf(al - mf[d]));
}

__global__ void k_edge_agg2(const int* __restrict__ src, const int* __restrict__ dst,
                            const float* __restrict__ as, const float* __restrict__ ad,
                            const float* __restrict__ mf, const float* __restrict__ s2,
                            const float* __restrict__ H2, float* __restrict__ OUT2) {
  int e = blockIdx.x * blockDim.x + threadIdx.x;
  if (e >= NEDGES) return;
  int s = src[e], d = dst[e];
  float al = lrelu(as[s] + ad[d]);
  float coef = __expf(al - mf[d]) / (s2[d] + GAT_EPS);
  F16 hp;
#pragma unroll
  for (int q = 0; q < 4; ++q)
    hp.v[q] = *(const float4*)(H2 + (size_t)s * NCLS + q * 4);
  float* op = OUT2 + (size_t)d * NCLS;
#pragma unroll
  for (int c = 0; c < NCLS; ++c) atomicAdd(op + c, hp.f[c] * coef);
}

__global__ void k_final(const float* __restrict__ H2, const float* __restrict__ as,
                        const float* __restrict__ ad, const float* __restrict__ mf,
                        const float* __restrict__ s2, const float* __restrict__ OUT2,
                        const float* __restrict__ bias2, float* __restrict__ out) {
  int n = blockIdx.x * blockDim.x + threadIdx.x;
  if (n >= NNODES) return;
  float self = lrelu(as[n] + ad[n]);
  float coef = __expf(self - mf[n]) / (s2[n] + GAT_EPS);
  F16 o2, h2, b;
#pragma unroll
  for (int q = 0; q < 4; ++q) {
    o2.v[q] = *(const float4*)(OUT2 + (size_t)n * NCLS + q * 4);
    h2.v[q] = *(const float4*)(H2 + (size_t)n * NCLS + q * 4);
    b.v[q]  = *(const float4*)(bias2 + q * 4);
  }
  F16 v; F16 lsm;
  float mx = -3.402823466e38f;
#pragma unroll
  for (int c = 0; c < NCLS; ++c) {
    v.f[c] = o2.f[c] + h2.f[c] * coef + b.f[c];
    mx = fmaxf(mx, v.f[c]);
  }
  float se = 0.f;
#pragma unroll
  for (int c = 0; c < NCLS; ++c) se += __expf(v.f[c] - mx);
  float lse = __logf(se) + mx;
#pragma unroll
  for (int c = 0; c < NCLS; ++c) lsm.f[c] = v.f[c] - lse;
  float4* o0 = (float4*)(out + (size_t)n * NCLS);
  float4* o1 = (float4*)(out + (size_t)NNODES * NCLS + (size_t)n * NCLS);
#pragma unroll
  for (int q = 0; q < 4; ++q) { o0[q] = v.v[q]; o1[q] = lsm.v[q]; }
}

// ---------------------------------------------------------------------------
extern "C" void kernel_launch(void* const* d_in, const int* in_sizes, int n_in,
                              void* d_out, int out_size, void* d_ws, size_t ws_size,
                              hipStream_t stream) {
  (void)in_sizes; (void)n_in; (void)out_size; (void)ws_size;
  const float* X   = (const float*)d_in[0];
  const int*   ei  = (const int*)d_in[1];
  const int*   src = ei;
  const int*   dst = ei + NEDGES;
  const float* W1  = (const float*)d_in[2];
  const float* aS1 = (const float*)d_in[3];
  const float* aD1 = (const float*)d_in[4];
  const float* b1  = (const float*)d_in[5];
  const float* W2  = (const float*)d_in[6];
  const float* aS2 = (const float*)d_in[7];
  const float* aD2 = (const float*)d_in[8];
  const float* b2  = (const float*)d_in[9];
  float* out = (float*)d_out;

  float* ws = (float*)d_ws;
  size_t off = 0;
  float* H1   = ws + off; off += (size_t)NNODES * C1;
  float* AS1  = ws + off; off += (size_t)NNODES * HEADS;
  float* AD1  = ws + off; off += (size_t)NNODES * HEADS;
  float* M1   = ws + off; off += (size_t)NNODES * HEADS;    // uint ord, then float
  float* S1   = ws + off; off += (size_t)NNODES * HEADS;
  float* OUT1 = ws + off; off += (size_t)NNODES * C1;       // agg -> ELU(features)
  float* H2   = ws + off; off += (size_t)NNODES * NCLS;
  float* AS2  = ws + off; off += (size_t)NNODES;
  float* AD2  = ws + off; off += (size_t)NNODES;
  float* M2   = ws + off; off += (size_t)NNODES;
  float* S2   = ws + off; off += (size_t)NNODES;
  float* OUT2 = ws + off; off += (size_t)NNODES * NCLS;

  const int B = 256;
  const int gMblk = (NNODES + 255) / 256;                 // 196 (WMMA blocks, 32 rows/wave)
  const int gE    = (NEDGES + B - 1) / B;                 // 6250
  const int gEH   = (NEDGES * HEADS + B - 1) / B;         // 50000
  const int gNH   = (NNODES * HEADS + B - 1) / B;
  const int gNC   = (NNODES * C1 + B - 1) / B;
  const int gN    = (NNODES + B - 1) / B;

  // zero accumulators with b128 stores (ordf domain: 0 < ord(x) for finite x)
  k_zero4<<<(NNODES * C1 / 4 + B - 1) / B, B, 0, stream>>>((float4*)OUT1, NNODES * C1 / 4);
  k_zero4<<<(NNODES * HEADS / 4 + B - 1) / B, B, 0, stream>>>((float4*)M1, NNODES * HEADS / 4);
  k_zero4<<<(NNODES / 4 + B - 1) / B, B, 0, stream>>>((float4*)M2, NNODES / 4);
  k_zero4<<<(NNODES * NCLS / 4 + B - 1) / B, B, 0, stream>>>((float4*)OUT2, NNODES * NCLS / 4);

  // ---- layer 1 ----
  k_gemm1<<<gMblk, B, 0, stream>>>(X, W1, H1);
  k_scores1<<<gNH, B, 0, stream>>>(H1, aS1, aD1, AS1, AD1);
  k_edge_max1<<<gE, B, 0, stream>>>(src, dst, AS1, AD1, (unsigned*)M1);
  k_node1<<<gNH, B, 0, stream>>>(AS1, AD1, (unsigned*)M1, S1);
  k_edge_sum1<<<gE, B, 0, stream>>>(src, dst, AS1, AD1, M1, S1);
  k_edge_agg1<<<gEH, B, 0, stream>>>(src, dst, AS1, AD1, M1, S1, H1, OUT1);
  k_out1<<<gNC, B, 0, stream>>>(AS1, AD1, M1, S1, H1, b1, OUT1);

  // ---- layer 2 ----
  k_gemm2<<<gMblk, B, 0, stream>>>(OUT1, W2, H2);
  k_scores2<<<gN, B, 0, stream>>>(H2, aS2, aD2, AS2, AD2);
  k_edge_max2<<<gE, B, 0, stream>>>(src, dst, AS2, AD2, (unsigned*)M2);
  k_node2<<<gN, B, 0, stream>>>(AS2, AD2, (unsigned*)M2, S2);
  k_edge_sum2<<<gE, B, 0, stream>>>(src, dst, AS2, AD2, M2, S2);
  k_edge_agg2<<<gE, B, 0, stream>>>(src, dst, AS2, AD2, M2, S2, H2, OUT2);
  k_final<<<gN, B, 0, stream>>>(H2, AS2, AD2, M2, S2, OUT2, b2, out);
}
